// SuperRes_82042465289115
// MI455X (gfx1250) — compile-verified, matrix operand order checked by reading, not run on hardware
//
#include <hip/hip_runtime.h>
#include <math.h>

// ---------------- constants ----------------
#define NPTS 15872
#define MPTS 35637
#define KNNB 5
#define KUPB 20

// Flip to 0 if the integrated assembler rejects the async mnemonics.
#define USE_ASYNC_LDS 1

typedef __attribute__((ext_vector_type(16))) _Float16 v16h;
typedef __attribute__((ext_vector_type(8)))  _Float16 v8h;
typedef __attribute__((ext_vector_type(8)))  float    v8f;

// Epilogue modes
#define EP_BIAS  0   // y + shift            (shift may be null)
#define EP_LRELU 1   // lrelu(y*scale+shift)
#define EP_SIN   2   // sin(y+shift)
#define EP_SINW  4   // sin(omega*(y+shift))

// Per the 16-bit 16x32 fragment layout (ISA 7.12.2): for a lane with
// half = lane>>4, the 16 fragment elements are two contiguous 8-half runs
// starting at k = off and k = off+16 (off = kc*32 + half*8).
// Row-major-K LDS tiles make each run one ds_load_b128.
__device__ __forceinline__ v16h load_frag(const _Float16* rowbase, int off) {
  v8h lo = *(const v8h*)(rowbase + off);
  v8h hi = *(const v8h*)(rowbase + off + 16);
  return __builtin_shufflevector(lo, hi, 0, 1, 2, 3, 4, 5, 6, 7,
                                 8, 9, 10, 11, 12, 13, 14, 15);
}

// Async global->LDS 16B transfer (GLOBAL_LOAD_ASYNC_TO_LDS_B128, ASYNCcnt).
// VDST carries the LDS byte address, VADDR the 64-bit global address (GV mode).
__device__ __forceinline__ void async_copy16(const void* g, void* lds) {
#if USE_ASYNC_LDS
  unsigned l = (unsigned)(uintptr_t)lds;
  unsigned long long ga = (unsigned long long)(uintptr_t)g;
  asm volatile("global_load_async_to_lds_b128 %0, %1, off"
               :: "v"(l), "v"(ga) : "memory");
#else
  *(uint4*)lds = *(const uint4*)g;
#endif
}
__device__ __forceinline__ void async_wait0() {
#if USE_ASYNC_LDS
  asm volatile("s_wait_asynccnt 0" ::: "memory");
#endif
}

// ---------------------------------------------------------------------------
// Generic WMMA GEMM: C[Mr,Nc] = epilogue(A[Mr,K] (f32) x Bh (f16,[Nc][K]))
// Block: 256 threads = 8 waves; tile 32 rows x 64 cols; K chunks of 32.
// ---------------------------------------------------------------------------
__global__ void gemm_wmma(const float* __restrict__ A,
                          const _Float16* __restrict__ Bh,  // [Nc][K] (o-major)
                          float* __restrict__ C,
                          int Mr, int Nc, int K,
                          const float* __restrict__ scale,
                          const float* __restrict__ shift,
                          int mode, float omega) {
  __shared__ __align__(16) _Float16 As[32][32];
  __shared__ __align__(16) _Float16 Bs[64][32];   // [col][k]
  const int tid  = threadIdx.x;
  const int wave = tid >> 5;
  const int lane = tid & 31;
  const int rh   = wave >> 2;         // row half 0/1
  const int cw   = wave & 3;          // col strip 0..3
  const int row0 = blockIdx.y * 32;
  const int colB = blockIdx.x * 64;
  const int m  = lane & 15;
  const int hf = lane >> 4;
  v8f acc = {};
  for (int k0 = 0; k0 < K; k0 += 32) {
    const bool fullA = (row0 + 32 <= Mr) && (k0 + 32 <= K);
    const bool fullB = (colB + 64 <= Nc) && (k0 + 32 <= K);
    if (fullB) {
      // 64x32 halves = 256 x 16B chunks; async DMA straight into LDS.
      for (int e = tid; e < 256; e += 256) {
        int cc = e >> 2, ch = (e & 3) * 8;
        async_copy16(&Bh[(size_t)(colB + cc) * K + k0 + ch], &Bs[cc][ch]);
      }
    } else {
      for (int e = tid; e < 64 * 32; e += 256) {
        int cc = e >> 5, kk = e & 31;
        _Float16 v = (_Float16)0.f;
        if (colB + cc < Nc && k0 + kk < K) v = Bh[(size_t)(colB + cc) * K + k0 + kk];
        Bs[cc][kk] = v;
      }
    }
    if (fullA) {
      // 32x32 halves = 512 f32 pairs; converting path stays on VGPRs.
      for (int e = tid; e < 512; e += 256) {
        int r = e >> 4, kk2 = (e & 15) * 2;
        const float* ap = &A[(size_t)(row0 + r) * K + k0 + kk2];
        union { _Float16 h[2]; unsigned u; } pk;
        pk.h[0] = (_Float16)ap[0];
        pk.h[1] = (_Float16)ap[1];
        *(unsigned*)&As[r][kk2] = pk.u;
      }
    } else {
      for (int e = tid; e < 32 * 32; e += 256) {
        int r = e >> 5, kk = e & 31;
        float v = 0.f;
        if (row0 + r < Mr && k0 + kk < K) v = A[(size_t)(row0 + r) * K + k0 + kk];
        As[r][kk] = (_Float16)v;
      }
    }
    if (k0 + 32 < K && row0 + (tid >> 3) < Mr)
      __builtin_prefetch(&A[(size_t)(row0 + (tid >> 3)) * K + k0 + 32], 0, 1);
    if (fullB) async_wait0();
    __syncthreads();
    v16h a = load_frag(&As[rh * 16 + m][0], hf * 8);
    v16h b = load_frag(&Bs[cw * 16 + m][0], hf * 8);
    acc = __builtin_amdgcn_wmma_f32_16x16x32_f16(false, a, false, b,
                                                 (short)0, acc, false, false);
    __syncthreads();
  }
#pragma unroll
  for (int r = 0; r < 8; ++r) {
    int gr = row0 + rh * 16 + r + 8 * hf;
    int gc = colB + cw * 16 + m;
    if (gr < Mr && gc < Nc) {
      float y = acc[r];
      float s = scale ? scale[gc] : 1.f;
      float t = shift ? shift[gc] : 0.f;
      if (mode == EP_LRELU)      { y = y * s + t; y = y > 0.f ? y : 0.2f * y; }
      else if (mode == EP_SIN)   { y = sinf(y + t); }
      else if (mode == EP_SINW)  { y = sinf(omega * (y + t)); }
      else                       { y = y + t; }
      C[(size_t)gr * Nc + gc] = y;
    }
  }
}

// ---------------------------------------------------------------------------
// Feature-space KNN: 16x16 Gram tiles via WMMA (K=64), streaming top-6.
// Block = 1 wave; tiles are contiguous 2KB row blocks of F, staged by
// async global->LDS DMA; both fragments are K-contiguous ds_load_b128s.
// ---------------------------------------------------------------------------
__global__ void knn_topk(const _Float16* __restrict__ F,   // [N][64]
                         const float* __restrict__ xx,     // [N]
                         int* __restrict__ outIdx) {       // [N][KNNB]
  __shared__ __align__(16) _Float16 As[16][64];
  __shared__ __align__(16) _Float16 Bs[16][64];   // [n][k]
  __shared__ float G[16][16];
  const int lane = threadIdx.x;
  const int row0 = blockIdx.x * 16;
  const int m  = lane & 15;
  const int hf = lane >> 4;
  {
    const uint4* s = (const uint4*)(F + (size_t)row0 * 64);
    uint4* d = (uint4*)&As[0][0];
    for (int e = lane; e < 128; e += 32) async_copy16(s + e, d + e);
  }
  float bv[6]; int bi[6];
#pragma unroll
  for (int i = 0; i < 6; ++i) { bv[i] = -3.4e38f; bi[i] = 0; }
  float xr = (lane < 16) ? xx[row0 + lane] : 0.f;
  for (int c0 = 0; c0 < NPTS; c0 += 16) {
    __syncthreads();
    {
      const uint4* s = (const uint4*)(F + (size_t)c0 * 64);
      uint4* d = (uint4*)&Bs[0][0];
      for (int e = lane; e < 128; e += 32) async_copy16(s + e, d + e);
    }
    async_wait0();
    __syncthreads();
    v8f acc = {};
#pragma unroll
    for (int kc = 0; kc < 2; ++kc) {
      v16h a = load_frag(&As[m][0], kc * 32 + hf * 8);
      v16h b = load_frag(&Bs[m][0], kc * 32 + hf * 8);
      acc = __builtin_amdgcn_wmma_f32_16x16x32_f16(false, a, false, b,
                                                   (short)0, acc, false, false);
    }
#pragma unroll
    for (int r = 0; r < 8; ++r) G[r + 8 * hf][m] = acc[r];
    __syncthreads();
    if (lane < 16) {
#pragma unroll
      for (int j = 0; j < 16; ++j) {
        float g = 2.f * G[lane][j] - xr - xx[c0 + j];   // -dist^2
        if (g > bv[5]) {
          int p = 5;
          while (p > 0 && g > bv[p - 1]) { bv[p] = bv[p - 1]; bi[p] = bi[p - 1]; --p; }
          bv[p] = g; bi[p] = c0 + j;
        }
      }
    }
  }
  if (lane < 16)
    for (int j = 0; j < KNNB; ++j)
      outIdx[(size_t)(row0 + lane) * KNNB + j] = bi[j + 1];  // drop self
}

// ---------------------------------------------------------------------------
// Fused 8-layer SIREN upscale stack: 16-row tiles, h in LDS (f16),
// per-layer 32KB weights DMA'd to LDS, 8 waves x 16-col WMMA strips.
// ---------------------------------------------------------------------------
__global__ void upscale_fused(const float* __restrict__ up_in,   // [R][7]
                              const float* __restrict__ W0,      // [128][7]
                              const float* __restrict__ s0p,
                              const float* __restrict__ h0p,
                              const _Float16* __restrict__ Wh,   // [7][128][128] o-major
                              const float* __restrict__ srest,   // [7][128]
                              const float* __restrict__ hrest,   // [7][128]
                              const float* __restrict__ wf,      // [128]
                              const float* __restrict__ bf,      // [1]
                              float* __restrict__ logits, int R) {
  __shared__ __align__(16) _Float16 Ws[128][128];  // [o][k]
  __shared__ __align__(16) _Float16 h_a[16][128];
  __shared__ __align__(16) _Float16 h_b[16][128];
  const int tid  = threadIdx.x;        // 256
  const int wave = tid >> 5;
  const int lane = tid & 31;
  const int row0 = blockIdx.x * 16;
  const int m  = lane & 15;
  const int hf = lane >> 4;
  // layer 0: 7 -> 128 (VALU, tiny K)
  for (int p = tid; p < 16 * 128; p += 256) {
    int o = p & 127, rr = p >> 7;
    int gr = row0 + rr;
    float y = 0.f;
    if (gr < R) {
      float acc = 0.f;
#pragma unroll
      for (int c = 0; c < 7; ++c) acc += up_in[(size_t)gr * 7 + c] * W0[o * 7 + c];
      y = sinf(acc * s0p[o] + h0p[o]);
    }
    h_a[rr][o] = (_Float16)y;
  }
  __syncthreads();
  _Float16 (*hc)[128] = h_a;
  _Float16 (*hn)[128] = h_b;
  for (int L = 0; L < 7; ++L) {
    {  // stage weights: 32KB async DMA (layout already [o][k])
      const uint4* s = (const uint4*)(Wh + (size_t)L * 128 * 128);
      uint4* d = (uint4*)&Ws[0][0];
      for (int e = tid; e < 2048; e += 256) async_copy16(s + e, d + e);
      async_wait0();
    }
    __syncthreads();
    v8f acc = {};
#pragma unroll
    for (int kc = 0; kc < 4; ++kc) {
      v16h a = load_frag(&hc[m][0], kc * 32 + hf * 8);
      v16h b = load_frag(&Ws[wave * 16 + m][0], kc * 32 + hf * 8);
      acc = __builtin_amdgcn_wmma_f32_16x16x32_f16(false, a, false, b,
                                                   (short)0, acc, false, false);
    }
    const float* sl = srest + L * 128;
    const float* hl = hrest + L * 128;
#pragma unroll
    for (int r = 0; r < 8; ++r) {
      int mm = r + 8 * hf;
      int o  = wave * 16 + m;
      hn[mm][o] = (_Float16)sinf(acc[r] * sl[o] + hl[o]);
    }
    __syncthreads();
    _Float16 (*t)[128] = hc; hc = hn; hn = t;
  }
  if (tid < 16) {
    int gr = row0 + tid;
    if (gr < R) {
      float acc = bf[0];
      for (int c = 0; c < 128; ++c) acc += (float)hc[tid][c] * wf[c];
      logits[gr] = acc;
    }
  }
}

// ---------------- glue kernels (bandwidth-bound VALU) ----------------------
__global__ void k_copy_cols(const float* __restrict__ src, float* __restrict__ dst,
                            int n, int Cs, int Cd, int off) {
  size_t t = (size_t)blockIdx.x * blockDim.x + threadIdx.x;
  if (t >= (size_t)n * Cs) return;
  int c = (int)(t % Cs); size_t r = t / Cs;
  dst[r * Cd + off + c] = src[t];
}
__global__ void k_bcast_cols(const float* __restrict__ vec, float* __restrict__ dst,
                             int n, int Cs, int Cd, int off) {
  size_t t = (size_t)blockIdx.x * blockDim.x + threadIdx.x;
  if (t >= (size_t)n * Cs) return;
  int c = (int)(t % Cs); size_t r = t / Cs;
  dst[r * Cd + off + c] = vec[c];
}
__global__ void k_cvt_f16(const float* __restrict__ x, _Float16* __restrict__ y, size_t n) {
  size_t t = (size_t)blockIdx.x * blockDim.x + threadIdx.x;
  if (t < n) y[t] = (_Float16)x[t];
}
__global__ void k_rowsq(const float* __restrict__ X, float* __restrict__ xx, int n, int C) {
  int r = blockIdx.x * blockDim.x + threadIdx.x;
  if (r >= n) return;
  float s = 0.f;
  for (int c = 0; c < C; ++c) { float v = X[(size_t)r * C + c]; s += v * v; }
  xx[r] = s;
}
__global__ void k_build_edge(const float* __restrict__ feat, const int* __restrict__ idx,
                             float* __restrict__ E, int n, int C, int Kn) {
  size_t t = (size_t)blockIdx.x * blockDim.x + threadIdx.x;
  size_t total = (size_t)n * Kn * 2 * C;
  if (t >= total) return;
  int c = (int)(t % (2 * C));
  size_t nk = t / (2 * C);
  int k = (int)(nk % Kn); int p = (int)(nk / Kn);
  int j = idx[(size_t)p * Kn + k];
  float ctr = feat[(size_t)p * C + (c < C ? c : c - C)];
  E[t] = (c < C) ? (feat[(size_t)j * C + c] - ctr) : ctr;
}
__global__ void k_max_over_k(const float* __restrict__ E, float* __restrict__ out,
                             int n, int C, int Kn) {
  size_t t = (size_t)blockIdx.x * blockDim.x + threadIdx.x;
  if (t >= (size_t)n * C) return;
  int c = (int)(t % C); size_t p = t / C;
  float mv = -3.4e38f;
  for (int k = 0; k < Kn; ++k) mv = fmaxf(mv, E[((size_t)p * Kn + k) * C + c]);
  out[t] = mv;
}
__global__ void k_col_reduce(const float* __restrict__ X, float* __restrict__ cmax,
                             float* __restrict__ cavg, int n, int C) {
  __shared__ float smax[256], ssum[256];
  int c = blockIdx.x, tid = threadIdx.x;
  float mv = -3.4e38f, s = 0.f;
  for (int r = tid; r < n; r += blockDim.x) {
    float v = X[(size_t)r * C + c];
    mv = fmaxf(mv, v); s += v;
  }
  smax[tid] = mv; ssum[tid] = s;
  __syncthreads();
  for (int st = 128; st > 0; st >>= 1) {
    if (tid < st) { smax[tid] = fmaxf(smax[tid], smax[tid + st]); ssum[tid] += ssum[tid + st]; }
    __syncthreads();
  }
  if (tid == 0) { cmax[c] = smax[0]; cavg[c] = ssum[0] / (float)n; }
}
__global__ void k_softmax_k(const float* __restrict__ logits, float* __restrict__ w, int Mp) {
  int mi = blockIdx.x * blockDim.x + threadIdx.x;
  if (mi >= Mp) return;
  float mx = -3.4e38f;
  for (int k = 0; k < KUPB; ++k) mx = fmaxf(mx, logits[(size_t)mi * KUPB + k]);
  float s = 0.f;
  for (int k = 0; k < KUPB; ++k) s += expf(logits[(size_t)mi * KUPB + k] - mx);
  float inv = 1.f / s;
  for (int k = 0; k < KUPB; ++k)
    w[(size_t)mi * KUPB + k] = expf(logits[(size_t)mi * KUPB + k] - mx) * inv;
}
__global__ void k_zgather(const float* __restrict__ w, const int* __restrict__ uidx,
                          const float* __restrict__ xf, float* __restrict__ z, int Mp) {
  size_t t = (size_t)blockIdx.x * blockDim.x + threadIdx.x;
  if (t >= (size_t)Mp * 227) return;
  int c = (int)(t % 227); size_t mi = t / 227;
  float acc = 0.f;
  for (int k = 0; k < KUPB; ++k) {
    int j = uidx[mi * KUPB + k];
    acc += w[mi * KUPB + k] * xf[(size_t)j * 227 + c];
  }
  z[t] = acc;
}

// ---------------------------------------------------------------------------
extern "C" void kernel_launch(void* const* d_in, const int* in_sizes, int n_in,
                              void* d_out, int out_size, void* d_ws, size_t ws_size,
                              hipStream_t stream) {
  (void)in_sizes; (void)n_in; (void)out_size; (void)ws_size;
#define IN_F(i) ((const float*)d_in[i])
#define IN_I(i) ((const int*)d_in[i])
  const float* ldx     = IN_F(1);
  const float* pos_emb = IN_F(2);
  const int*   idx0    = IN_I(3);
  const float *b0_ec_w1 = IN_F(4),  *b0_ec_s1 = IN_F(5),  *b0_ec_h1 = IN_F(6);
  const float *b0_ec_w2 = IN_F(7),  *b0_ec_b2 = IN_F(8);
  const float *b0_fc_w1 = IN_F(9),  *b0_fc_s1 = IN_F(10), *b0_fc_h1 = IN_F(11);
  const float *b0_fc_w2 = IN_F(12), *b0_fc_s2 = IN_F(13), *b0_fc_h2 = IN_F(14);
  const float *b0_fc_w3 = IN_F(15), *b0_fc_s3 = IN_F(16), *b0_fc_h3 = IN_F(17);
  const float *b0_fc_w4 = IN_F(18), *b0_fc_b4 = IN_F(19);
  const float *b1_ec_w1 = IN_F(20), *b1_ec_s1 = IN_F(21), *b1_ec_h1 = IN_F(22);
  const float *b1_ec_w2 = IN_F(23), *b1_ec_b2 = IN_F(24);
  const float *b1_fc_w1 = IN_F(25), *b1_fc_s1 = IN_F(26), *b1_fc_h1 = IN_F(27);
  const float *b1_fc_w2 = IN_F(28), *b1_fc_s2 = IN_F(29), *b1_fc_h2 = IN_F(30);
  const float *b1_fc_w3 = IN_F(31), *b1_fc_s3 = IN_F(32), *b1_fc_h3 = IN_F(33);
  const float *b1_fc_w4 = IN_F(34), *b1_fc_b4 = IN_F(35);
  const float *up_inputs = IN_F(36);
  const int   *up_idx    = IN_I(37);
  const float *up_w0 = IN_F(38), *up_s0 = IN_F(39), *up_h0 = IN_F(40);
  const float *up_w_rest = IN_F(41), *up_s_rest = IN_F(42), *up_h_rest = IN_F(43);
  const float *up_wf = IN_F(44), *up_bf = IN_F(45);
  const float *dec_w0 = IN_F(46), *dec_b0 = IN_F(47);
  const float *dec_w1 = IN_F(48), *dec_b1 = IN_F(49);
  const float *dec_w2 = IN_F(50), *dec_b2 = IN_F(51);
  const float *dec_w3 = IN_F(52), *dec_b3 = IN_F(53);
  const float *dec_w4 = IN_F(54), *dec_b4 = IN_F(55);
  const float *dec_wf = IN_F(56), *dec_bf = IN_F(57);
  float* outp = (float*)d_out;

  // bump allocator over workspace
  char* base = (char*)d_ws;
  size_t off = 0;
  auto allocB = [&](size_t bytes) -> void* {
    void* p = base + off;
    off = (off + bytes + 255) & ~(size_t)255;
    return p;
  };
  auto allocF = [&](size_t n) { return (float*)allocB(n * 4); };
  auto allocH = [&](size_t n) { return (_Float16*)allocB(n * 2); };
  auto allocI = [&](size_t n) { return (int*)allocB(n * 4); };

  const int N = NPTS, M = MPTS;
  const int R = M * KUPB;

  // f16 weights in native [O][C] (= [col][k]) layout -- no transpose needed.
  _Float16 *tb0e1 = allocH(128 * 70),  *tb0e2 = allocH(64 * 128);
  _Float16 *tb0f1 = allocH(192 * 227), *tb0f2 = allocH(128 * 192);
  _Float16 *tb0f3 = allocH(64 * 128),  *tb0f4 = allocH(64 * 64);
  _Float16 *tb1e1 = allocH(256 * 128), *tb1e2 = allocH(128 * 256);
  _Float16 *tb1f1 = allocH(384 * 448), *tb1f2 = allocH(256 * 384);
  _Float16 *tb1f3 = allocH(128 * 256), *tb1f4 = allocH(128 * 128);
  _Float16 *tupW  = allocH(7 * 128 * 128);
  _Float16 *td0 = allocH(256 * 227), *td1 = allocH(128 * 256), *td2 = allocH(64 * 128);
  _Float16 *td3 = allocH(32 * 64),   *td4 = allocH(16 * 32),   *tdf = allocH(3 * 16);
  // activations
  float *ft_in = allocF((size_t)N * 35);
  float *E0 = allocF((size_t)N * KNNB * 70);
  float *t1 = allocF((size_t)N * KNNB * 128);
  float *t2 = allocF((size_t)N * KNNB * 64);
  float *f1l0 = allocF((size_t)N * 64);
  float *cmax0 = allocF(64), *cavg0 = allocF(64);
  float *H0 = allocF((size_t)N * 227);
  float *g1 = allocF((size_t)N * 192), *g2 = allocF((size_t)N * 128);
  float *g3 = allocF((size_t)N * 64),  *f0 = allocF((size_t)N * 64);
  _Float16 *f0h = allocH((size_t)N * 64);
  float *xx = allocF(N);
  int *idx1 = allocI((size_t)N * KNNB);
  float *E1 = allocF((size_t)N * KNNB * 128);
  float *u1 = allocF((size_t)N * KNNB * 256);
  float *u2 = allocF((size_t)N * KNNB * 128);
  float *f1l1 = allocF((size_t)N * 128);
  float *cmax1 = allocF(128), *cavg1 = allocF(128);
  float *H1 = allocF((size_t)N * 448);
  float *q1 = allocF((size_t)N * 384), *q2 = allocF((size_t)N * 256);
  float *q3 = allocF((size_t)N * 128), *f1 = allocF((size_t)N * 128);
  float *xfeat = allocF((size_t)N * 227);
  float *logits = allocF(R);
  float *wts = allocF(R);
  float *z = allocF((size_t)M * 227);
  float *dh1 = allocF((size_t)M * 256), *dh2 = allocF((size_t)M * 128);
  float *dh3 = allocF((size_t)M * 64),  *dh4 = allocF((size_t)M * 32);
  float *dh5 = allocF((size_t)M * 16);

  auto g1d = [](size_t tot) { return dim3((unsigned)((tot + 255) / 256)); };
  auto CV = [&](const float* W, _Float16* Wh, size_t n) {
    k_cvt_f16<<<g1d(n), 256, 0, stream>>>(W, Wh, n);
  };
  auto GEMM = [&](const float* A, const _Float16* Bh, float* C, int Mr, int Nc,
                  int K, const float* sc, const float* sh, int mode, float om) {
    dim3 g((unsigned)((Nc + 63) / 64), (unsigned)((Mr + 31) / 32));
    gemm_wmma<<<g, 256, 0, stream>>>(A, Bh, C, Mr, Nc, K, sc, sh, mode, om);
  };

  // --- weight prep: f32 -> f16, keep native [O][C] layout ---
  CV(b0_ec_w1, tb0e1, 128 * 70);   CV(b0_ec_w2, tb0e2, 64 * 128);
  CV(b0_fc_w1, tb0f1, 192 * 227);  CV(b0_fc_w2, tb0f2, 128 * 192);
  CV(b0_fc_w3, tb0f3, 64 * 128);   CV(b0_fc_w4, tb0f4, 64 * 64);
  CV(b1_ec_w1, tb1e1, 256 * 128);  CV(b1_ec_w2, tb1e2, 128 * 256);
  CV(b1_fc_w1, tb1f1, 384 * 448);  CV(b1_fc_w2, tb1f2, 256 * 384);
  CV(b1_fc_w3, tb1f3, 128 * 256);  CV(b1_fc_w4, tb1f4, 128 * 128);
  CV(up_w_rest, tupW, 7 * 128 * 128);
  CV(dec_w0, td0, 256 * 227); CV(dec_w1, td1, 128 * 256); CV(dec_w2, td2, 64 * 128);
  CV(dec_w3, td3, 32 * 64);   CV(dec_w4, td4, 16 * 32);   CV(dec_wf, tdf, 3 * 16);

  // --- feat_in = [ldx | pos_emb] ---
  k_copy_cols<<<g1d((size_t)N * 3), 256, 0, stream>>>(ldx, ft_in, N, 3, 35, 0);
  k_copy_cols<<<g1d((size_t)N * 32), 256, 0, stream>>>(pos_emb, ft_in, N, 32, 35, 3);

  // --- EmbModule 0 ---
  k_build_edge<<<g1d((size_t)N * KNNB * 70), 256, 0, stream>>>(ft_in, idx0, E0, N, 35, KNNB);
  GEMM(E0, tb0e1, t1, N * KNNB, 128, 70, b0_ec_s1, b0_ec_h1, EP_LRELU, 0.f);
  GEMM(t1, tb0e2, t2, N * KNNB, 64, 128, nullptr, b0_ec_b2, EP_BIAS, 0.f);
  k_max_over_k<<<g1d((size_t)N * 64), 256, 0, stream>>>(t2, f1l0, N, 64, KNNB);
  k_col_reduce<<<64, 256, 0, stream>>>(f1l0, cmax0, cavg0, N, 64);
  k_copy_cols<<<g1d((size_t)N * 35), 256, 0, stream>>>(ft_in, H0, N, 35, 227, 0);
  k_copy_cols<<<g1d((size_t)N * 64), 256, 0, stream>>>(f1l0, H0, N, 64, 227, 35);
  k_bcast_cols<<<g1d((size_t)N * 64), 256, 0, stream>>>(cmax0, H0, N, 64, 227, 99);
  k_bcast_cols<<<g1d((size_t)N * 64), 256, 0, stream>>>(cavg0, H0, N, 64, 227, 163);
  GEMM(H0, tb0f1, g1, N, 192, 227, b0_fc_s1, b0_fc_h1, EP_LRELU, 0.f);
  GEMM(g1, tb0f2, g2, N, 128, 192, b0_fc_s2, b0_fc_h2, EP_LRELU, 0.f);
  GEMM(g2, tb0f3, g3, N, 64, 128, b0_fc_s3, b0_fc_h3, EP_LRELU, 0.f);
  GEMM(g3, tb0f4, f0, N, 64, 64, nullptr, b0_fc_b4, EP_BIAS, 0.f);

  // --- dynamic feature-space KNN via WMMA Gram tiles ---
  k_cvt_f16<<<g1d((size_t)N * 64), 256, 0, stream>>>(f0, f0h, (size_t)N * 64);
  k_rowsq<<<g1d(N), 256, 0, stream>>>(f0, xx, N, 64);
  knn_topk<<<N / 16, 32, 0, stream>>>(f0h, xx, idx1);

  // --- EmbModule 1 ---
  k_build_edge<<<g1d((size_t)N * KNNB * 128), 256, 0, stream>>>(f0, idx1, E1, N, 64, KNNB);
  GEMM(E1, tb1e1, u1, N * KNNB, 256, 128, b1_ec_s1, b1_ec_h1, EP_LRELU, 0.f);
  GEMM(u1, tb1e2, u2, N * KNNB, 128, 256, nullptr, b1_ec_b2, EP_BIAS, 0.f);
  k_max_over_k<<<g1d((size_t)N * 128), 256, 0, stream>>>(u2, f1l1, N, 128, KNNB);
  k_col_reduce<<<128, 256, 0, stream>>>(f1l1, cmax1, cavg1, N, 128);
  k_copy_cols<<<g1d((size_t)N * 64), 256, 0, stream>>>(f0, H1, N, 64, 448, 0);
  k_copy_cols<<<g1d((size_t)N * 128), 256, 0, stream>>>(f1l1, H1, N, 128, 448, 64);
  k_bcast_cols<<<g1d((size_t)N * 128), 256, 0, stream>>>(cmax1, H1, N, 128, 448, 192);
  k_bcast_cols<<<g1d((size_t)N * 128), 256, 0, stream>>>(cavg1, H1, N, 128, 448, 320);
  GEMM(H1, tb1f1, q1, N, 384, 448, b1_fc_s1, b1_fc_h1, EP_LRELU, 0.f);
  GEMM(q1, tb1f2, q2, N, 256, 384, b1_fc_s2, b1_fc_h2, EP_LRELU, 0.f);
  GEMM(q2, tb1f3, q3, N, 128, 256, b1_fc_s3, b1_fc_h3, EP_LRELU, 0.f);
  GEMM(q3, tb1f4, f1, N, 128, 128, nullptr, b1_fc_b4, EP_BIAS, 0.f);

  // --- x_feat = [feat_in | f0 | f1] ---
  k_copy_cols<<<g1d((size_t)N * 35), 256, 0, stream>>>(ft_in, xfeat, N, 35, 227, 0);
  k_copy_cols<<<g1d((size_t)N * 64), 256, 0, stream>>>(f0, xfeat, N, 64, 227, 35);
  k_copy_cols<<<g1d((size_t)N * 128), 256, 0, stream>>>(f1, xfeat, N, 128, 227, 99);

  // --- fused SIREN upscale stack -> interpolation logits ---
  upscale_fused<<<(R + 15) / 16, 256, 0, stream>>>(up_inputs, up_w0, up_s0, up_h0,
                                                   tupW, up_s_rest, up_h_rest,
                                                   up_wf, up_bf, logits, R);
  k_softmax_k<<<g1d(M), 256, 0, stream>>>(logits, wts, M);
  k_zgather<<<g1d((size_t)M * 227), 256, 0, stream>>>(wts, up_idx, xfeat, z, M);

  // --- SIREN decoder ---
  GEMM(z, td0, dh1, M, 256, 227, nullptr, dec_b0, EP_SINW, 30.f);
  GEMM(dh1, td1, dh2, M, 128, 256, nullptr, dec_b1, EP_SIN, 0.f);
  GEMM(dh2, td2, dh3, M, 64, 128, nullptr, dec_b2, EP_SIN, 0.f);
  GEMM(dh3, td3, dh4, M, 32, 64, nullptr, dec_b3, EP_SIN, 0.f);
  GEMM(dh4, td4, dh5, M, 16, 32, nullptr, dec_b4, EP_SIN, 0.f);
  GEMM(dh5, tdf, outp, M, 3, 16, nullptr, dec_bf, EP_BIAS, 0.f);
#undef IN_F
#undef IN_I
}
